// SculptedContrastiveLoss_81235011437122
// MI455X (gfx1250) — matrix operand order repaired and checked by reading, not compiled, not run on hardware
//
#include <hip/hip_runtime.h>

typedef __attribute__((ext_vector_type(16))) __bf16 v16bf;
typedef __attribute__((ext_vector_type(8)))  float  v8f;

#define BATCH 8192
#define DIM   512
#define TILE  128
#define KT    32
#define KSTEPS (DIM / KT)
#define LDK   (KT + 2)   // pad LDS rows to stagger banks

__device__ __forceinline__ __bf16 to_bf16(float f) { return (__bf16)f; }

// ---------------------------------------------------------------------------
// Kernel 1: per-row sq = sum(e^2), s = sum(e), plus attractive-term partials.
// One wave (32 lanes) per row; 8 rows per 256-thread block.
// ---------------------------------------------------------------------------
__global__ __launch_bounds__(256) void row_stats(
    const float* __restrict__ E, const int* __restrict__ lab,
    const float* __restrict__ centers,
    float* __restrict__ sq, float* __restrict__ s, float* __restrict__ attAcc)
{
    const int lane = threadIdx.x & 31;
    const int wave = threadIdx.x >> 5;
    const int row  = blockIdx.x * 8 + wave;

    const float* e = E + (size_t)row * DIM;
    const float* c = centers + (size_t)lab[row] * DIM;

    float ssq = 0.f, ss = 0.f, att = 0.f;
    #pragma unroll 4
    for (int d = lane; d < DIM; d += 32) {
        const float v = e[d];
        const float diff = v - c[d];
        ssq = fmaf(v, v, ssq);
        ss += v;
        att = fmaf(diff, diff, att);
    }
    #pragma unroll
    for (int off = 16; off > 0; off >>= 1) {
        ssq += __shfl_down(ssq, off, 32);
        ss  += __shfl_down(ss,  off, 32);
        att += __shfl_down(att, off, 32);
    }
    if (lane == 0) {
        sq[row] = ssq;
        s[row]  = ss;
        atomicAdd(attAcc, att);
    }
}

// ---------------------------------------------------------------------------
// Kernel 2: tiled Gram (bf16 WMMA, fp32 accumulate) + hinge epilogue.
// Block = 256 threads = 8 waves; block tile 128x128; wave tile 32x64
// (2 M-tiles x 4 N-tiles of 16x16). Upper-triangle blocks only.
// ---------------------------------------------------------------------------
__global__ __launch_bounds__(256) void gram_hinge(
    const float* __restrict__ E, const int* __restrict__ lab,
    const float* __restrict__ sq, const float* __restrict__ s,
    float* __restrict__ hingeAcc)
{
    const int bi = blockIdx.y;   // i block (rows)
    const int bj = blockIdx.x;   // j block (cols)
    if (bj < bi) return;         // block-uniform: EXEC stays all-ones for WMMA

    __shared__ __align__(16) __bf16 Asm[TILE][LDK];
    __shared__ __align__(16) __bf16 Bsm[TILE][LDK];
    __shared__ float wred[8];

    const int t     = threadIdx.x;
    const int lane  = t & 31;
    const int wave  = t >> 5;
    const int waveM = wave & 3;   // 4 wave-rows of 32
    const int waveN = wave >> 2;  // 2 wave-cols of 64
    const int l15   = lane & 15;
    const int hi    = (lane >> 4) & 1;

    // Cooperative tile loading: thread -> (row lr in 0..31, float4 col lc4)
    const int lr = t >> 3;
    const int lc = (t & 7) * 4;

    v8f acc[2][4] = {};

    const float* Abase = E + (size_t)(bi * TILE) * DIM;
    const float* Bbase = E + (size_t)(bj * TILE) * DIM;

    for (int kt = 0; kt < KSTEPS; ++kt) {
        __syncthreads();   // previous iteration's fragment reads done
        const int kcol = kt * KT + lc;
        #pragma unroll
        for (int p = 0; p < 4; ++p) {
            const int row = p * 32 + lr;
            const float4 av = *reinterpret_cast<const float4*>(Abase + (size_t)row * DIM + kcol);
            const float4 bv = *reinterpret_cast<const float4*>(Bbase + (size_t)row * DIM + kcol);
            Asm[row][lc + 0] = to_bf16(av.x); Asm[row][lc + 1] = to_bf16(av.y);
            Asm[row][lc + 2] = to_bf16(av.z); Asm[row][lc + 3] = to_bf16(av.w);
            Bsm[row][lc + 0] = to_bf16(bv.x); Bsm[row][lc + 1] = to_bf16(bv.y);
            Bsm[row][lc + 2] = to_bf16(bv.z); Bsm[row][lc + 3] = to_bf16(bv.w);
            if (kt + 1 < KSTEPS) {   // global_prefetch_b8 for next k-tile
                __builtin_prefetch(Abase + (size_t)row * DIM + kcol + KT, 0, 1);
                __builtin_prefetch(Bbase + (size_t)row * DIM + kcol + KT, 0, 1);
            }
        }
        __syncthreads();

        // A fragment, 16-bit 16x32 layout (ISA 7.12.2):
        //   lanes 0-15: row=l15, K pairs {0,2,4,6,16,18,20,22}
        //   lanes 16-31: same rows, K pairs +8
        v16bf afr[2];
        #pragma unroll
        for (int mt = 0; mt < 2; ++mt) {
            const int ar   = waveM * 32 + mt * 16 + l15;
            const int aoff = hi ? 8 : 0;
            #pragma unroll
            for (int v = 0; v < 8; ++v) {
                const int k = 2 * v + ((v >= 4) ? 8 : 0) + aoff;
                afr[mt][2 * v]     = Asm[ar][k];
                afr[mt][2 * v + 1] = Asm[ar][k + 1];
            }
        }
        // B fragment, 32x16 (K x N): lane n holds N=l15;
        //   lanes 0-15: K=0..7,16..23 packed 2/VGPR; lanes 16-31: K=8..15,24..31.
        // Bsm stores j-rows of E (N x K), so B[k][n] = Bsm[n][k].
        v16bf bfr[4];
        #pragma unroll
        for (int nt = 0; nt < 4; ++nt) {
            const int bn   = waveN * 64 + nt * 16 + l15;
            const int boff = hi ? 8 : 0;
            #pragma unroll
            for (int v = 0; v < 8; ++v) {
                const int k = 2 * v + ((v >= 4) ? 8 : 0) + boff;
                bfr[nt][2 * v]     = Bsm[bn][k];
                bfr[nt][2 * v + 1] = Bsm[bn][k + 1];
            }
        }
        #pragma unroll
        for (int mt = 0; mt < 2; ++mt)
            #pragma unroll
            for (int nt = 0; nt < 4; ++nt)
                acc[mt][nt] = __builtin_amdgcn_wmma_f32_16x16x32_bf16(
                    false, afr[mt], false, bfr[nt],
                    (short)0, acc[mt][nt], false, false);
    }

    // Epilogue: C/D layout -> VGPR r: lanes 0-15 M=r, N=lane; lanes 16-31 M=r+8.
    float local = 0.f;
    const int iBase = bi * TILE + waveM * 32 + hi * 8;
    const int jBase = bj * TILE + waveN * 64 + l15;
    #pragma unroll
    for (int nt = 0; nt < 4; ++nt) {
        const int j = jBase + nt * 16;
        const float sqj = sq[j];
        const float sj  = s[j];
        const int   lj  = lab[j];
        #pragma unroll
        for (int mt = 0; mt < 2; ++mt) {
            #pragma unroll
            for (int r = 0; r < 8; ++r) {
                const int i = iBase + mt * 16 + r;
                if (i < j && lab[i] != lj) {
                    const float g = acc[mt][nt][r];
                    float d2 = sq[i] + sqj - 2.0f * g
                             + 2.0e-6f * (s[i] - sj) + (float)DIM * 1.0e-12f;
                    d2 = fmaxf(d2, 0.0f);
                    const float h = fmaxf(0.3f - __builtin_sqrtf(d2), 0.0f);
                    local = fmaf(h, h, local);
                }
            }
        }
    }
    #pragma unroll
    for (int off = 16; off > 0; off >>= 1) local += __shfl_down(local, off, 32);
    if (lane == 0) wred[wave] = local;
    __syncthreads();
    if (t == 0) {
        float tot = 0.f;
        #pragma unroll
        for (int w = 0; w < 8; ++w) tot += wred[w];
        atomicAdd(hingeAcc, tot);
    }
}

// ---------------------------------------------------------------------------
__global__ void init_acc(float* acc)
{
    if (threadIdx.x < 2) acc[threadIdx.x] = 0.f;
}

__global__ void finalize(const float* __restrict__ acc, float* __restrict__ out)
{
    if (threadIdx.x == 0) {
        const float attractive = acc[0] / ((float)BATCH * (float)DIM);
        const float num_pairs  = 0.5f * (float)BATCH * (float)(BATCH - 1);
        out[0] = attractive + acc[1] / num_pairs;
    }
}

// ---------------------------------------------------------------------------
extern "C" void kernel_launch(void* const* d_in, const int* in_sizes, int n_in,
                              void* d_out, int out_size, void* d_ws, size_t ws_size,
                              hipStream_t stream)
{
    const float* E       = (const float*)d_in[0];
    const int*   lab     = (const int*)d_in[1];
    const float* centers = (const float*)d_in[2];

    float* ws  = (float*)d_ws;
    float* acc = ws;            // [0] = attractive sum, [1] = hinge sum
    float* sq  = ws + 16;       // [BATCH]
    float* s   = ws + 16 + BATCH; // [BATCH]

    hipLaunchKernelGGL(init_acc, dim3(1), dim3(32), 0, stream, acc);
    hipLaunchKernelGGL(row_stats, dim3(BATCH / 8), dim3(256), 0, stream,
                       E, lab, centers, sq, s, acc + 0);
    hipLaunchKernelGGL(gram_hinge, dim3(BATCH / TILE, BATCH / TILE), dim3(256), 0, stream,
                       E, lab, sq, s, acc + 1);
    hipLaunchKernelGGL(finalize, dim3(1), dim3(32), 0, stream, acc, (float*)d_out);
}
